// TemporalFusionTransformer_90323162235391
// MI455X (gfx1250) — compile-verified, matrix operand order checked by reading, not run on hardware
//
#include <hip/hip_runtime.h>
#include <hip/hip_bf16.h>
#include <hip/hip_fp16.h>
#include <math.h>

typedef _Float16 h16;
typedef __attribute__((ext_vector_type(16))) _Float16 v16h;
typedef __attribute__((ext_vector_type(8)))  _Float16 v8h;
typedef __attribute__((ext_vector_type(8)))  float    v8f;
typedef __attribute__((ext_vector_type(4)))  float    v4f;

#define DEV __device__ __forceinline__

// ---------------- problem constants ----------------
constexpr int  Bk  = 128;          // batch
constexpr int  Sq  = 256;          // sequence
constexpr int  Fi  = 8;            // features
constexpr int  Hd  = 256;          // hidden
constexpr int  NHd = 4;            // heads
constexpr int  DAd = 64;           // head dim
constexpr int  NT  = Bk * Sq;      // 32768 tokens
constexpr int  FH  = Fi * Hd;      // 2048
constexpr long SLOT = (long)NT * Hd;       // 8,388,608 floats
constexpr long BIG  = (long)NT * FH;       // 67,108,864 floats
constexpr long ATTN_SZ = (long)Bk * NHd * Sq * Sq; // 33,554,432
constexpr float EPSV = 1e-5f;

// workspace layout (floats)
constexpr long O_EMB = 0;
constexpr long O_B1  = O_EMB + BIG;
constexpr long O_B2  = O_B1 + BIG;
constexpr long O_SEL = O_B2 + BIG;
constexpr long O_T8  = O_SEL + SLOT;
constexpr long O_R8  = O_T8 + (long)NT * 8;
constexpr long O_AVG = O_R8 + (long)NT * 8;
constexpr long O_HID = O_AVG + (long)Bk * Sq * Sq;
constexpr long O_F16 = O_HID + (long)Bk * 128;   // f16 weight arena starts here

// ---------------- device helpers ----------------
DEV float sigm(float x) { return 1.f / (1.f + __expf(-x)); }

DEV float wsum(float v) {
#pragma unroll
  for (int m = 16; m >= 1; m >>= 1) v += __shfl_xor(v, m, 32);
  return v;
}
DEV float wmaxr(float v) {
#pragma unroll
  for (int m = 16; m >= 1; m >>= 1) v = fmaxf(v, __shfl_xor(v, m, 32));
  return v;
}

// A/B fragment for v_wmma_f32_16x16x32_f16:
// lane half (lane>>4) selects K windows [kc+8*half, +8) and [kc+16+8*half, +8)
DEV v16h frag_ld(const float* __restrict__ row, int kc, int half) {
  const v4f* p0 = (const v4f*)(row + kc + half * 8);
  const v4f* p1 = (const v4f*)(row + kc + 16 + half * 8);
  v4f a0 = p0[0], a1 = p0[1], b0 = p1[0], b1 = p1[1];
  v16h r;
#pragma unroll
  for (int i = 0; i < 4; i++) {
    r[i]      = (h16)a0[i];
    r[4 + i]  = (h16)a1[i];
    r[8 + i]  = (h16)b0[i];
    r[12 + i] = (h16)b1[i];
  }
  return r;
}
DEV v16h frag_ld(const h16* __restrict__ row, int kc, int half) {
  v8h a = *(const v8h*)(row + kc + half * 8);
  v8h b = *(const v8h*)(row + kc + 16 + half * 8);
  v16h r;
#pragma unroll
  for (int i = 0; i < 8; i++) { r[i] = a[i]; r[8 + i] = b[i]; }
  return r;
}

DEV v8f wmma32(v16h a, v16h b, v8f c) {
  return __builtin_amdgcn_wmma_f32_16x16x32_f16(false, a, false, b, (short)0, c,
                                                false, false);
}

// ---------------- generic WMMA GEMM ----------------
// C[rows x M] = A[rows x K] @ W^T (W stored pre-transposed f16, Mpad rows x K cols)
// EPI: 0 none, 1 ELU, 2 ReLU. Batched via gridDim.z with strides sA/sW/sC.
template <int EPI>
__global__ void gemm_k(const float* __restrict__ A, long lda, long sA,
                       const h16* __restrict__ W, long ldw, long sW,
                       const float* __restrict__ bias,
                       float* __restrict__ C, long ldc, long sC,
                       int rows, int K, int M) {
  int z = blockIdx.z;
  A += (long)z * sA; W += (long)z * sW; C += (long)z * sC;
  int lane = threadIdx.x & 31, wave = threadIdx.x >> 5;
  int rowbase = blockIdx.y * 64 + wave * 16;
  if (rowbase >= rows) return;
  int colbase = blockIdx.x * 64;
  int half = lane >> 4;
  const float* arow = A + (long)(rowbase + (lane & 15)) * lda;
  v8f acc[4] = {};
  for (int kc = 0; kc < K; kc += 32) {
    v16h a = frag_ld(arow, kc, half);
#pragma unroll
    for (int c = 0; c < 4; c++) {
      int n = colbase + c * 16 + (lane & 15);
      v16h b = frag_ld(W + (long)n * ldw, kc, half);
      acc[c] = wmma32(a, b, acc[c]);
    }
  }
#pragma unroll
  for (int c = 0; c < 4; c++) {
    int n = colbase + c * 16 + (lane & 15);
    if (n >= M) continue;
    float bb = bias ? bias[n] : 0.f;
#pragma unroll
    for (int r = 0; r < 8; r++) {
      int row = rowbase + r + half * 8;
      if (row >= rows) continue;
      float v = acc[c][r] + bb;
      if (EPI == 1) v = v > 0.f ? v : (__expf(v) - 1.f);
      if (EPI == 2) v = fmaxf(v, 0.f);
      C[(long)row * ldc + n] = v;
    }
  }
}

// fused GLU GEMM: C = sigmoid(A@Wg + bg) * (A@Wv + bv); per-wave tile 16x32
__global__ void glugemm_k(const float* __restrict__ A, long lda,
                          const h16* __restrict__ Wg, const h16* __restrict__ Wv,
                          long ldw, const float* __restrict__ bg,
                          const float* __restrict__ bv, float* __restrict__ C,
                          long ldc, int rows, int K, int M) {
  int lane = threadIdx.x & 31, wave = threadIdx.x >> 5;
  int rowbase = blockIdx.y * 64 + wave * 16;
  if (rowbase >= rows) return;
  int colbase = blockIdx.x * 32;
  int half = lane >> 4;
  const float* arow = A + (long)(rowbase + (lane & 15)) * lda;
  v8f ag[2] = {}, av[2] = {};
  for (int kc = 0; kc < K; kc += 32) {
    v16h a = frag_ld(arow, kc, half);
#pragma unroll
    for (int c = 0; c < 2; c++) {
      int n = colbase + c * 16 + (lane & 15);
      v16h bgf = frag_ld(Wg + (long)n * ldw, kc, half);
      v16h bvf = frag_ld(Wv + (long)n * ldw, kc, half);
      ag[c] = wmma32(a, bgf, ag[c]);
      av[c] = wmma32(a, bvf, av[c]);
    }
  }
#pragma unroll
  for (int c = 0; c < 2; c++) {
    int n = colbase + c * 16 + (lane & 15);
    if (n >= M) continue;
    float gb = bg[n], vb2 = bv[n];
#pragma unroll
    for (int r = 0; r < 8; r++) {
      int row = rowbase + r + half * 8;
      if (row >= rows) continue;
      C[(long)row * ldc + n] = sigm(ag[c][r] + gb) * (av[c][r] + vb2);
    }
  }
}

// attention scores: attn[(b,h,q,k)] = 0.125 * q_bh @ k_bh^T  (K=DA=64, B operand f32)
__global__ void scores_k(const float* __restrict__ q, const float* __restrict__ k,
                         float* __restrict__ attn) {
  int z = blockIdx.z; int b = z >> 2; int h = z & 3;
  const float* A = q + (long)b * Sq * Hd + h * DAd;
  const float* W = k + (long)b * Sq * Hd + h * DAd;
  float* C = attn + (long)z * Sq * Sq;
  int lane = threadIdx.x & 31, wave = threadIdx.x >> 5;
  int rowbase = blockIdx.y * 64 + wave * 16;
  int colbase = blockIdx.x * 64;
  int half = lane >> 4;
  const float* arow = A + (long)(rowbase + (lane & 15)) * Hd;
  v8f acc[4] = {};
  for (int kc = 0; kc < DAd; kc += 32) {
    v16h a = frag_ld(arow, kc, half);
#pragma unroll
    for (int c = 0; c < 4; c++) {
      int n = colbase + c * 16 + (lane & 15);
      v16h bf = frag_ld(W + (long)n * Hd, kc, half);
      acc[c] = wmma32(a, bf, acc[c]);
    }
  }
#pragma unroll
  for (int c = 0; c < 4; c++) {
    int n = colbase + c * 16 + (lane & 15);
#pragma unroll
    for (int r = 0; r < 8; r++) {
      int row = rowbase + r + half * 8;
      C[(long)row * Sq + n] = 0.125f * acc[c][r];
    }
  }
}

// ---------------- elementwise / reduction kernels ----------------
__global__ void embed_k(const float* __restrict__ x, const float* __restrict__ w,
                        const float* __restrict__ b, float* __restrict__ emb) {
  long i = (long)blockIdx.x * 256 + threadIdx.x;
  if (i >= BIG) return;
  int h = (int)(i & (Hd - 1));
  long nf = i >> 8;
  emb[i] = x[nf] * w[h] + b[h];
}

// out[row] = LN(x[row] + y[row]) * g + b  over C=256; gamma row = (row % gmod)
__global__ void ln_add_k(const float* __restrict__ x, const float* __restrict__ y,
                         const float* __restrict__ g, const float* __restrict__ b,
                         float* __restrict__ out, long rows, int gmod) {
  int lane = threadIdx.x & 31, wave = threadIdx.x >> 5;
  long row = (long)blockIdx.x * 8 + wave;
  if (row >= rows) return;
  const float* xr = x + row * 256;
  const float* yr = y + row * 256;
  float v[8]; float s = 0.f;
#pragma unroll
  for (int i = 0; i < 8; i++) { v[i] = xr[lane + i * 32] + yr[lane + i * 32]; s += v[i]; }
  float mu = wsum(s) * (1.f / 256.f);
  float s2 = 0.f;
#pragma unroll
  for (int i = 0; i < 8; i++) { float d = v[i] - mu; s2 += d * d; }
  float rstd = rsqrtf(wsum(s2) * (1.f / 256.f) + EPSV);
  long go = gmod ? (long)(row % gmod) * 256 : 0;
#pragma unroll
  for (int i = 0; i < 8; i++) {
    int c = lane + i * 32;
    out[row * 256 + c] = (v[i] - mu) * rstd * g[go + c] + b[go + c];
  }
}

__global__ void softmax_k(float* __restrict__ a, long rows) {
  int lane = threadIdx.x & 31, wave = threadIdx.x >> 5;
  long row = (long)blockIdx.x * 8 + wave;
  if (row >= rows) return;
  float* r = a + row * 256;
  float v[8]; float m = -1e30f;
#pragma unroll
  for (int i = 0; i < 8; i++) { v[i] = r[lane + i * 32]; m = fmaxf(m, v[i]); }
  m = wmaxr(m);
  float s = 0.f;
#pragma unroll
  for (int i = 0; i < 8; i++) { v[i] = __expf(v[i] - m); s += v[i]; }
  float inv = 1.f / wsum(s);
#pragma unroll
  for (int i = 0; i < 8; i++) r[lane + i * 32] = v[i] * inv;
}

// VSN weight GRN tail: 8x8 GLU + LN(8) + softmax(8), one token per thread
__global__ void vsn_small_k(const float* __restrict__ t8, const float* __restrict__ r8,
                            const float* __restrict__ wg, const float* __restrict__ bg,
                            const float* __restrict__ wv, const float* __restrict__ bv,
                            const float* __restrict__ lng, const float* __restrict__ lnb,
                            float* __restrict__ varw) {
  int n = blockIdx.x * 256 + threadIdx.x;
  if (n >= NT) return;
  float tt[8], s[8];
#pragma unroll
  for (int j = 0; j < 8; j++) tt[j] = t8[(long)n * 8 + j];
  float mu = 0.f;
#pragma unroll
  for (int i = 0; i < 8; i++) {
    float g = bg[i], v = bv[i];
#pragma unroll
    for (int j = 0; j < 8; j++) { g += tt[j] * wg[j * 8 + i]; v += tt[j] * wv[j * 8 + i]; }
    s[i] = r8[(long)n * 8 + i] + sigm(g) * v;
    mu += s[i];
  }
  mu *= 0.125f;
  float var = 0.f;
#pragma unroll
  for (int i = 0; i < 8; i++) { float d = s[i] - mu; var += d * d; }
  float rstd = rsqrtf(var * 0.125f + EPSV);
  float mx = -1e30f;
#pragma unroll
  for (int i = 0; i < 8; i++) { s[i] = (s[i] - mu) * rstd * lng[i] + lnb[i]; mx = fmaxf(mx, s[i]); }
  float sum = 0.f;
#pragma unroll
  for (int i = 0; i < 8; i++) { s[i] = __expf(s[i] - mx); sum += s[i]; }
  float inv = 1.f / sum;
#pragma unroll
  for (int i = 0; i < 8; i++) varw[(long)n * 8 + i] = s[i] * inv;
}

// selected[n,h] = sum_f processed[(n*8+f),h] * varw[n*8+f]
__global__ void selected_k(const float* __restrict__ proc, const float* __restrict__ vw,
                           float* __restrict__ sel) {
  long i = (long)blockIdx.x * 256 + threadIdx.x;
  if (i >= SLOT) return;
  long n = i >> 8; int h = (int)(i & 255);
  float s = 0.f;
#pragma unroll
  for (int f = 0; f < 8; f++)
    s += proc[((n << 3) + f) * 256 + h] * vw[(n << 3) + f];
  sel[i] = s;
}

// full LSTM scan in ONE launch: block b owns batch element b for all 256 steps.
// Hidden state double-buffered in LDS (one barrier per step); cell state in regs.
__global__ void lstm_k(const float* __restrict__ xw, const float* __restrict__ whh,
                       const float* __restrict__ bhh, float* __restrict__ lstm_out) {
  int b = blockIdx.x, j = threadIdx.x;
  __shared__ float hs[2][256];
  hs[0][j] = 0.f;
  float c = 0.f;
  float bi = bhh[j], bf = bhh[256 + j], bg = bhh[512 + j], bo = bhh[768 + j];
  __syncthreads();
  for (int t = 0; t < Sq; t++) {
    const float* hcur = hs[t & 1];
    const float* xr = xw + ((long)b * 256 + t) * 1024;
    float gi = xr[j] + bi;
    float gf = xr[256 + j] + bf;
    float gg = xr[512 + j] + bg;
    float go = xr[768 + j] + bo;
    for (int k = 0; k < 256; k++) {
      float h = hcur[k];
      const float* w = whh + (long)k * 1024;
      gi += h * w[j]; gf += h * w[256 + j]; gg += h * w[512 + j]; go += h * w[768 + j];
    }
    c = sigm(gf) * c + sigm(gi) * tanhf(gg);
    float h = sigm(go) * tanhf(c);
    hs[(t + 1) & 1][j] = h;
    lstm_out[((long)b * 256 + t) * 256 + j] = h;
    __syncthreads();
  }
}

__global__ void avgheads_k(const float* __restrict__ attn, float* __restrict__ avg) {
  long i = (long)blockIdx.x * 256 + threadIdx.x;
  if (i >= (long)Bk * Sq * Sq) return;
  long b = i / ((long)Sq * Sq);
  long r = i % ((long)Sq * Sq);
  const float* base = attn + b * NHd * (long)Sq * Sq + r;
  avg[i] = 0.25f * (base[0] + base[(long)Sq * Sq] + base[2L * Sq * Sq] + base[3L * Sq * Sq]);
}

// vt[b,d,s] = (h16) v[b,s,d]
__global__ void vt_k(const float* __restrict__ v, h16* __restrict__ vt) {
  long i = (long)blockIdx.x * 256 + threadIdx.x;
  if (i >= (long)Bk * DAd * Sq) return;
  long b = i / ((long)DAd * Sq);
  long rem = i % ((long)DAd * Sq);
  long d = rem / Sq, s = rem % Sq;
  vt[i] = (h16)v[(b * Sq + s) * DAd + d];
}

// weight prep: Wt[n,k] (f16, Mpad rows) = W[k,n] (f32), zero-padded rows
__global__ void prep_wt_k(const float* __restrict__ W, h16* __restrict__ Wt, int K,
                          int M, long total) {
  long i = (long)blockIdx.x * 256 + threadIdx.x;
  if (i >= total) return;
  long n = i / K, k = i % K;
  Wt[i] = (n < M) ? (h16)W[k * (long)M + n] : (h16)0.f;
}

// ---------------- host launcher ----------------
extern "C" void kernel_launch(void* const* d_in, const int* in_sizes, int n_in,
                              void* d_out, int out_size, void* d_ws, size_t ws_size,
                              hipStream_t stream) {
  (void)in_sizes; (void)n_in; (void)out_size; (void)ws_size;
  auto in = [&](int i) { return (const float*)d_in[i]; };
  float* out = (float*)d_out;

  float* wsf = (float*)d_ws;
  float* emb = wsf + O_EMB;
  float* b1  = wsf + O_B1;
  float* b2  = wsf + O_B2;
  float* sel = wsf + O_SEL;
  float* t8  = wsf + O_T8;
  float* r8  = wsf + O_R8;
  float* avg = wsf + O_AVG;
  float* hid = wsf + O_HID;

  // output regions: (preds, attn_w, var_w) concatenated flat
  float* preds  = out;                       // 128*24*3
  float* attn_o = out + 9216;                // B*NH*S*S
  float* varw   = out + 9216 + ATTN_SZ;      // B*S*F

  // emb-region slots (free after `selected`)
  float* lstm_out    = emb + 0 * SLOT;
  float* temporal    = emb + 1 * SLOT;
  float* enriched    = emb + 2 * SLOT;
  float* qb          = emb + 3 * SLOT;
  float* kb          = emb + 4 * SLOT;
  float* vb          = emb + 5 * SLOT;
  float* ctx         = vb + (long)NT * DAd;
  float* attn_out    = emb + 6 * SLOT;
  float* attn_output = emb + 7 * SLOT;
  // b1-region slots (xW occupies slots 0-3 during the LSTM)
  float* xw  = b1;
  float* glu = b1 + 4 * SLOT;
  float* gh1 = b1 + 5 * SLOT;
  float* gh2 = b1 + 6 * SLOT;
  float* ffo = b1 + 7 * SLOT;

  // ---- f16 transposed-weight arena ----
  h16* wf = (h16*)(wsf + O_F16);
  long wofs = 0;
  auto walloc = [&](long halves) { h16* p = wf + wofs; wofs += halves; return p; };
  auto prep = [&](const float* W, int K, int M) {
    int Mpad = ((M + 63) / 64) * 64;
    long total = (long)Mpad * K;
    h16* Wt = walloc(total);
    prep_wt_k<<<dim3((unsigned)((total + 255) / 256)), dim3(256), 0, stream>>>(W, Wt, K, M, total);
    return Wt;
  };

  h16* w1g = prep(in(3), FH, Hd);
  h16* w2g = prep(in(5), Hd, 8);
  h16* wsg = prep(in(13), FH, 8);
  h16 *fw1[8], *fw2[8], *fwg[8], *fwv[8];
  for (int f = 0; f < 8; f++) fw1[f] = prep(in(15) + (long)f * Hd * Hd, Hd, Hd);
  for (int f = 0; f < 8; f++) fw2[f] = prep(in(17) + (long)f * Hd * Hd, Hd, Hd);
  for (int f = 0; f < 8; f++) fwg[f] = prep(in(19) + (long)f * Hd * Hd, Hd, Hd);
  for (int f = 0; f < 8; f++) fwv[f] = prep(in(21) + (long)f * Hd * Hd, Hd, Hd);
  h16* wih = prep(in(25), Hd, 4 * Hd);
  h16* pwg = prep(in(29), Hd, Hd);
  h16* pwv = prep(in(31), Hd, Hd);
  h16* ew1 = prep(in(35), Hd, Hd);
  h16* ew2 = prep(in(37), Hd, Hd);
  h16* ewg = prep(in(39), Hd, Hd);
  h16* ewv = prep(in(41), Hd, Hd);
  h16* wq  = prep(in(45), Hd, Hd);
  h16* wk  = prep(in(46), Hd, Hd);
  h16* wvv = prep(in(47), Hd, DAd);
  h16* wo  = prep(in(48), DAd, Hd);
  h16* agw = prep(in(50), Hd, Hd);
  h16* agv = prep(in(52), Hd, Hd);
  h16* fg1 = prep(in(56), Hd, Hd);
  h16* fg2 = prep(in(58), Hd, Hd);
  h16* fgg = prep(in(60), Hd, Hd);
  h16* fgv = prep(in(62), Hd, Hd);
  h16* o1  = prep(in(66), Hd, 128);
  h16* o2  = prep(in(68), 128, 72);
  h16* vt  = walloc((long)Bk * DAd * Sq);

  // ---- launch helpers ----
  auto gemm = [&](const float* A, long lda, const h16* W, int K, const float* bias,
                  float* C, long ldc, int rows, int M, int epi) {
    dim3 g((M + 63) / 64, (rows + 63) / 64, 1);
    if (epi == 0)
      gemm_k<0><<<g, dim3(128), 0, stream>>>(A, lda, 0, W, K, 0, bias, C, ldc, 0, rows, K, M);
    else if (epi == 1)
      gemm_k<1><<<g, dim3(128), 0, stream>>>(A, lda, 0, W, K, 0, bias, C, ldc, 0, rows, K, M);
    else
      gemm_k<2><<<g, dim3(128), 0, stream>>>(A, lda, 0, W, K, 0, bias, C, ldc, 0, rows, K, M);
  };
  auto glugemm = [&](const float* A, long lda, const h16* Wg, const h16* Wv, int K,
                     const float* bgp, const float* bvp, float* C, long ldc, int rows, int M) {
    dim3 g((M + 31) / 32, (rows + 63) / 64, 1);
    glugemm_k<<<g, dim3(128), 0, stream>>>(A, lda, Wg, Wv, K, bgp, bvp, C, ldc, rows, K, M);
  };
  auto ln_add = [&](const float* x, const float* y, const float* g, const float* b,
                    float* o, long rows, int gmod) {
    ln_add_k<<<dim3((unsigned)((rows + 7) / 8)), dim3(256), 0, stream>>>(x, y, g, b, o, rows, gmod);
  };

  // 1) embedding
  embed_k<<<dim3((unsigned)((BIG + 255) / 256)), dim3(256), 0, stream>>>(in(0), in(1), in(2), emb);

  // 2) VSN weight GRN -> var_w (written straight into d_out)
  gemm(emb, FH, w1g, FH, in(4), b1, Hd, NT, Hd, 1);          // wh1 = elu(flat@w1+b1)
  gemm(b1, Hd, w2g, Hd, in(6), t8, 8, NT, 8, 0);             // t8  = wh1@w2+b2
  gemm(emb, FH, wsg, FH, in(14), r8, 8, NT, 8, 0);           // r8  = flat@ws+bs
  vsn_small_k<<<dim3(NT / 256), dim3(256), 0, stream>>>(t8, r8, in(7), in(8), in(9),
                                                        in(10), in(11), in(12), varw);

  // 3) per-feature VSN GRNs
  for (int f = 0; f < 8; f++)
    gemm(emb + f * Hd, FH, fw1[f], Hd, in(16) + f * Hd, b1 + f * Hd, FH, NT, Hd, 1);
  for (int f = 0; f < 8; f++)
    gemm(b1 + f * Hd, FH, fw2[f], Hd, in(18) + f * Hd, b2 + f * Hd, FH, NT, Hd, 0);
  for (int f = 0; f < 8; f++)
    glugemm(b2 + f * Hd, FH, fwg[f], fwv[f], Hd, in(20) + f * Hd, in(22) + f * Hd,
            b1 + f * Hd, FH, NT, Hd);
  ln_add(emb, b1, in(23), in(24), b2, (long)NT * Fi, Fi);    // processed -> b2

  // 4) variable selection
  selected_k<<<dim3((unsigned)((SLOT + 255) / 256)), dim3(256), 0, stream>>>(b2, varw, sel);

  // 5) LSTM: one GEMM for the input projection, one persistent-scan launch
  gemm(sel, Hd, wih, Hd, in(27), xw, 4 * Hd, NT, 4 * Hd, 0); // xW = sel@w_ih + b_ih
  lstm_k<<<dim3(Bk), dim3(Hd), 0, stream>>>(xw, in(26), in(28), lstm_out);

  // 6) post-LSTM gate + LN
  glugemm(lstm_out, Hd, pwg, pwv, Hd, in(30), in(32), glu, Hd, NT, Hd);
  ln_add(sel, glu, in(33), in(34), temporal, NT, 0);

  // 7) enrichment GRN
  gemm(temporal, Hd, ew1, Hd, in(36), gh1, Hd, NT, Hd, 1);
  gemm(gh1, Hd, ew2, Hd, in(38), gh2, Hd, NT, Hd, 0);
  glugemm(gh2, Hd, ewg, ewv, Hd, in(40), in(42), glu, Hd, NT, Hd);
  ln_add(temporal, glu, in(43), in(44), enriched, NT, 0);

  // 8) attention
  gemm(enriched, Hd, wq, Hd, nullptr, qb, Hd, NT, Hd, 0);
  gemm(enriched, Hd, wk, Hd, nullptr, kb, Hd, NT, Hd, 0);
  gemm(enriched, Hd, wvv, Hd, nullptr, vb, DAd, NT, DAd, 0);
  scores_k<<<dim3(4, 4, Bk * NHd), dim3(128), 0, stream>>>(qb, kb, attn_o);
  softmax_k<<<dim3((unsigned)((Bk * NHd * Sq) / 8)), dim3(256), 0, stream>>>(
      attn_o, (long)Bk * NHd * Sq);
  avgheads_k<<<dim3((unsigned)(((long)Bk * Sq * Sq + 255) / 256)), dim3(256), 0, stream>>>(
      attn_o, avg);
  vt_k<<<dim3((unsigned)(((long)Bk * DAd * Sq + 255) / 256)), dim3(256), 0, stream>>>(vb, vt);
  // ctx[b] = avg[b] @ v[b]   (batched over grid.z)
  gemm_k<0><<<dim3(1, 4, Bk), dim3(128), 0, stream>>>(avg, Sq, (long)Sq * Sq, vt, Sq,
                                                      (long)DAd * Sq, nullptr, ctx, DAd,
                                                      (long)Sq * DAd, Sq, Sq, DAd);
  gemm(ctx, DAd, wo, DAd, in(49), attn_out, Hd, NT, Hd, 0);
  glugemm(attn_out, Hd, agw, agv, Hd, in(51), in(53), glu, Hd, NT, Hd);
  ln_add(enriched, glu, in(54), in(55), attn_output, NT, 0);

  // 9) feed-forward GRN
  gemm(attn_output, Hd, fg1, Hd, in(57), gh1, Hd, NT, Hd, 1);
  gemm(gh1, Hd, fg2, Hd, in(59), gh2, Hd, NT, Hd, 0);
  glugemm(gh2, Hd, fgg, fgv, Hd, in(61), in(63), glu, Hd, NT, Hd);
  ln_add(attn_output, glu, in(64), in(65), ffo, NT, 0);

  // 10) output head on last token
  gemm(ffo + 255L * Hd, (long)Sq * Hd, o1, Hd, in(67), hid, 128, Bk, 128, 2); // relu
  gemm(hid, 128, o2, 128, in(69), preds, 72, Bk, 72, 0);
}
